// DCPNet_34308198760857
// MI455X (gfx1250) — matrix operand order repaired and checked by reading, not compiled, not run on hardware
//
#include <hip/hip_runtime.h>
#include <math.h>

// Problem constants (from reference setup_inputs)
#define BB 16
#define NN 1024
#define DD 512
#define MM 1024

typedef __attribute__((ext_vector_type(16))) _Float16 v16h;
typedef __attribute__((ext_vector_type(8)))  _Float16 v8h;
typedef __attribute__((ext_vector_type(8)))  float    v8f;

// ---------------------------------------------------------------------------
// Kernel 1: [B,D,N] f32 -> [B,N,D] f16 (LDS tiled transpose + convert)
// ---------------------------------------------------------------------------
__global__ __launch_bounds__(256) void dcp_transpose_f16(
    const float* __restrict__ in, _Float16* __restrict__ out) {
  __shared__ float tile[32][33];
  const int b  = blockIdx.z;
  const int n0 = blockIdx.x * 32;
  const int d0 = blockIdx.y * 32;
  const int tx = threadIdx.x, ty = threadIdx.y;
#pragma unroll
  for (int i = 0; i < 4; ++i) {
    int d = d0 + ty + i * 8;
    tile[ty + i * 8][tx] = in[((size_t)b * DD + d) * NN + n0 + tx];
  }
  __syncthreads();
#pragma unroll
  for (int i = 0; i < 4; ++i) {
    int n = n0 + ty + i * 8;
    out[((size_t)b * NN + n) * DD + d0 + tx] = (_Float16)tile[tx][ty + i * 8];
  }
}

// ---------------------------------------------------------------------------
// Kernel 2: per-column sum of squares:  out[b*N+n] = sum_d in[b,d,n]^2 (f32)
// ---------------------------------------------------------------------------
__global__ __launch_bounds__(256) void dcp_sumsq(
    const float* __restrict__ in, float* __restrict__ out) {
  const int idx = blockIdx.x * 256 + threadIdx.x;   // b*NN + n
  const int b = idx >> 10, n = idx & 1023;
  const float* p = in + (size_t)b * DD * NN + n;
  float acc = 0.0f;
#pragma unroll 4
  for (int d = 0; d < DD; ++d) {
    float v = p[(size_t)d * NN];
    acc += v * v;
  }
  out[idx] = acc;
}

// ---------------------------------------------------------------------------
// Kernel 3: fused  inner-product (WMMA f16) -> euclid dist -> softmax ->
//           soft correspondence  src_corr[b,3,n]
// One block = one (b, 16-row n-block). 4 waves; wave w owns m in [256w,256w+256).
// Scores for the 16x1024 row block are staged in 64KB LDS, then softmaxed.
// ---------------------------------------------------------------------------
__global__ __launch_bounds__(128) void dcp_main(
    const _Float16* __restrict__ X16,   // [B,N,D] f16
    const _Float16* __restrict__ Y16,   // [B,M,D] f16
    const float* __restrict__ xx,       // [B,N]
    const float* __restrict__ yy,       // [B,M]
    const float* __restrict__ tgts,     // [B,3,M] f32
    float* __restrict__ corr)           // [B,3,N] f32 out
{
  __shared__ float sc[16 * 1024];       // exactly 64KB

  const int b    = blockIdx.x >> 6;
  const int n0   = (blockIdx.x & 63) * 16;
  const int tid  = threadIdx.x;
  const int wave = tid >> 5;
  const int lane = tid & 31;
  const int half = lane >> 4;           // 0: lanes 0-15, 1: lanes 16-31
  const int l15  = lane & 15;

  // ---- Phase 1: WMMA GEMM + distance scores into LDS ----
  // Preload xx for the 8 C-rows this lane-half owns (C VGPR r -> row r + 8*half)
  float xxr[8];
#pragma unroll
  for (int r = 0; r < 8; ++r)
    xxr[r] = xx[b * NN + n0 + r + half * 8];

  // A fragment source: row = n0 + l15, K base offset = half*8 (lo) / +16 (hi)
  const _Float16* arow = X16 + ((size_t)(b * NN + n0 + l15)) * DD + half * 8;

  for (int t = 0; t < 16; ++t) {
    const int m0 = (wave * 16 + t) * 16;
    // B fragment source: col = m0 + l15, K base offset = half*8 (lo) / +16 (hi)
    const _Float16* brow = Y16 + ((size_t)(b * MM + m0 + l15)) * DD + half * 8;

    v8f c = {};
#pragma unroll
    for (int k = 0; k < 16; ++k) {
      const v8h a_lo = *(const v8h*)(arow + k * 32);
      const v8h a_hi = *(const v8h*)(arow + k * 32 + 16);
      const v8h b_lo = *(const v8h*)(brow + k * 32);
      const v8h b_hi = *(const v8h*)(brow + k * 32 + 16);
      v16h av = __builtin_shufflevector(a_lo, a_hi, 0, 1, 2, 3, 4, 5, 6, 7,
                                        8, 9, 10, 11, 12, 13, 14, 15);
      v16h bv = __builtin_shufflevector(b_lo, b_hi, 0, 1, 2, 3, 4, 5, 6, 7,
                                        8, 9, 10, 11, 12, 13, 14, 15);
      c = __builtin_amdgcn_wmma_f32_16x16x32_f16(
          /*neg_a=*/false, av, /*neg_b=*/false, bv,
          /*c_mod=*/(short)0, c, /*reuse_a=*/false, /*reuse_b=*/false);
    }

    const int mIdx = m0 + l15;           // C column N = lane%16
    const float yym = yy[b * MM + mIdx];
#pragma unroll
    for (int r = 0; r < 8; ++r) {
      float pd = xxr[r] - 2.0f * c[r] + yym;
      float s = pd > 0.0f ? -sqrtf(pd) : 0.0f;   // score = -distance
      sc[(r + half * 8) * 1024 + mIdx] = s;
    }
  }
  __syncthreads();

  // ---- Phase 2: softmax over m + weighted sum with tgts -> src_corr ----
  const int row = tid >> 3;             // 0..15
  const int sub = tid & 7;              // 8 threads per row
  const float* srow = sc + row * 1024;

  float mx = -1e30f;
  for (int j = sub; j < 1024; j += 8) mx = fmaxf(mx, srow[j]);
#pragma unroll
  for (int o = 1; o < 8; o <<= 1) mx = fmaxf(mx, __shfl_xor(mx, o, 8));

  const float* tb = tgts + (size_t)b * 3 * MM;
  float se = 0.0f, a0 = 0.0f, a1 = 0.0f, a2 = 0.0f;
  for (int j = sub; j < 1024; j += 8) {
    float e = expf(srow[j] - mx);
    se += e;
    a0 += e * tb[j];
    a1 += e * tb[MM + j];
    a2 += e * tb[2 * MM + j];
  }
#pragma unroll
  for (int o = 1; o < 8; o <<= 1) {
    se += __shfl_xor(se, o, 8);
    a0 += __shfl_xor(a0, o, 8);
    a1 += __shfl_xor(a1, o, 8);
    a2 += __shfl_xor(a2, o, 8);
  }
  if (sub == 0) {
    const int n = n0 + row;
    const float inv = 1.0f / se;
    corr[((size_t)b * 3 + 0) * NN + n] = a0 * inv;
    corr[((size_t)b * 3 + 1) * NN + n] = a1 * inv;
    corr[((size_t)b * 3 + 2) * NN + n] = a2 * inv;
  }
}

// ---------------------------------------------------------------------------
// Kernel 4: per-batch rigid alignment (means, H, 3x3 SVD via Jacobi on H^T H,
// reflection fix, translation, sxyz Euler). One thread per batch.
// ---------------------------------------------------------------------------
__global__ __launch_bounds__(32) void dcp_finalize(
    const float* __restrict__ srcs,   // [B,3,N]
    const float* __restrict__ corr,   // [B,3,N]
    float* __restrict__ out)          // [B,6]
{
  const int bt = threadIdx.x;
  if (bt >= BB) return;
  const float* S = srcs + (size_t)bt * 3 * NN;
  const float* C = corr + (size_t)bt * 3 * NN;

  // means
  float sm[3] = {0, 0, 0}, cm[3] = {0, 0, 0};
  for (int n = 0; n < NN; ++n) {
#pragma unroll
    for (int i = 0; i < 3; ++i) {
      sm[i] += S[i * NN + n];
      cm[i] += C[i * NN + n];
    }
  }
#pragma unroll
  for (int i = 0; i < 3; ++i) { sm[i] *= (1.0f / NN); cm[i] *= (1.0f / NN); }

  // cross-covariance H[i][j] = sum_n (S[i,n]-sm[i])*(C[j,n]-cm[j])
  float H[3][3] = {{0, 0, 0}, {0, 0, 0}, {0, 0, 0}};
  for (int n = 0; n < NN; ++n) {
    float sv[3], cv[3];
#pragma unroll
    for (int i = 0; i < 3; ++i) {
      sv[i] = S[i * NN + n] - sm[i];
      cv[i] = C[i * NN + n] - cm[i];
    }
#pragma unroll
    for (int i = 0; i < 3; ++i)
#pragma unroll
      for (int j = 0; j < 3; ++j) H[i][j] += sv[i] * cv[j];
  }

  // A = H^T H (symmetric), Jacobi eigendecomposition: A = V diag(lam) V^T
  float A[3][3], V[3][3] = {{1, 0, 0}, {0, 1, 0}, {0, 0, 1}};
#pragma unroll
  for (int i = 0; i < 3; ++i)
#pragma unroll
    for (int j = 0; j < 3; ++j) {
      float a = 0.0f;
      for (int k = 0; k < 3; ++k) a += H[k][i] * H[k][j];
      A[i][j] = a;
    }

  auto rot = [&](int p, int q) {
    float apq = A[p][q];
    if (fabsf(apq) < 1e-30f) return;
    float tau = (A[q][q] - A[p][p]) / (2.0f * apq);
    float tt = (tau >= 0.0f ? 1.0f : -1.0f) /
               (fabsf(tau) + sqrtf(1.0f + tau * tau));
    float cc = 1.0f / sqrtf(1.0f + tt * tt);
    float ss = tt * cc;
    for (int k = 0; k < 3; ++k) {            // A <- A * J
      float akp = A[k][p], akq = A[k][q];
      A[k][p] = cc * akp - ss * akq;
      A[k][q] = ss * akp + cc * akq;
    }
    for (int k = 0; k < 3; ++k) {            // A <- J^T * A
      float apk = A[p][k], aqk = A[q][k];
      A[p][k] = cc * apk - ss * aqk;
      A[q][k] = ss * apk + cc * aqk;
    }
    for (int k = 0; k < 3; ++k) {            // V <- V * J
      float vkp = V[k][p], vkq = V[k][q];
      V[k][p] = cc * vkp - ss * vkq;
      V[k][q] = ss * vkp + cc * vkq;
    }
  };
  for (int sweep = 0; sweep < 12; ++sweep) { rot(0, 1); rot(0, 2); rot(1, 2); }

  float lam[3] = {A[0][0], A[1][1], A[2][2]};
  // sort eigenpairs descending (3-element)
  for (int i = 0; i < 2; ++i)
    for (int j = i + 1; j < 3; ++j)
      if (lam[j] > lam[i]) {
        float tl = lam[i]; lam[i] = lam[j]; lam[j] = tl;
        for (int k = 0; k < 3; ++k) {
          float tv = V[k][i]; V[k][i] = V[k][j]; V[k][j] = tv;
        }
      }

  // U[:,i] = H * V[:,i] / s_i
  float U[3][3];
  for (int i = 0; i < 3; ++i) {
    float si = sqrtf(fmaxf(lam[i], 0.0f));
    float inv = 1.0f / fmaxf(si, 1e-20f);
    for (int k = 0; k < 3; ++k) {
      float hv = 0.0f;
      for (int j = 0; j < 3; ++j) hv += H[k][j] * V[j][i];
      U[k][i] = hv * inv;
    }
  }

  // r = V U^T ; reflection fix on det < 0 (flip last column of V)
  float R[3][3];
  for (int i = 0; i < 3; ++i)
    for (int j = 0; j < 3; ++j) {
      float a = 0.0f;
      for (int k = 0; k < 3; ++k) a += V[i][k] * U[j][k];
      R[i][j] = a;
    }
  float det = R[0][0] * (R[1][1] * R[2][2] - R[1][2] * R[2][1]) -
              R[0][1] * (R[1][0] * R[2][2] - R[1][2] * R[2][0]) +
              R[0][2] * (R[1][0] * R[2][1] - R[1][1] * R[2][0]);
  if (det < 0.0f) {
    for (int k = 0; k < 3; ++k) V[k][2] = -V[k][2];
    for (int i = 0; i < 3; ++i)
      for (int j = 0; j < 3; ++j) {
        float a = 0.0f;
        for (int k = 0; k < 3; ++k) a += V[i][k] * U[j][k];
        R[i][j] = a;
      }
  }

  // t = -R * sm + cm
  float tv[3];
#pragma unroll
  for (int i = 0; i < 3; ++i) {
    float a = 0.0f;
    for (int j = 0; j < 3; ++j) a += R[i][j] * sm[j];
    tv[i] = cm[i] - a;
  }

  // sxyz Euler
  float cy = sqrtf(R[2][2] * R[2][2] + R[1][2] * R[1][2]);
  float ex = atan2f(-R[1][2], R[2][2]);
  float ey = atan2f(R[0][2], cy);
  float ez = atan2f(-R[0][1], R[0][0]);

  float* o = out + bt * 6;
  o[0] = ex; o[1] = ey; o[2] = ez;
  o[3] = tv[0]; o[4] = tv[1]; o[5] = tv[2];
}

// ---------------------------------------------------------------------------
extern "C" void kernel_launch(void* const* d_in, const int* in_sizes, int n_in,
                              void* d_out, int out_size, void* d_ws, size_t ws_size,
                              hipStream_t stream) {
  (void)in_sizes; (void)n_in; (void)out_size; (void)ws_size;
  const float* srcs = (const float*)d_in[0];   // [16,3,1024]
  const float* tgts = (const float*)d_in[1];   // [16,3,1024]
  const float* semb = (const float*)d_in[2];   // [16,512,1024]
  const float* temb = (const float*)d_in[3];   // [16,512,1024]
  float* out = (float*)d_out;                  // [16,6]

  // workspace layout (bytes): X16(16MB) | Y16(16MB) | xx | yy | corr
  char* ws = (char*)d_ws;
  const size_t embBytes = (size_t)BB * NN * DD * sizeof(_Float16); // 16 MB
  _Float16* X16 = (_Float16*)(ws);
  _Float16* Y16 = (_Float16*)(ws + embBytes);
  float* xx   = (float*)(ws + 2 * embBytes);
  float* yyb  = xx + BB * NN;
  float* corr = yyb + BB * MM;

  dim3 tb(32, 8), tg(NN / 32, DD / 32, BB);
  dcp_transpose_f16<<<tg, tb, 0, stream>>>(semb, X16);
  dcp_transpose_f16<<<tg, tb, 0, stream>>>(temb, Y16);
  dcp_sumsq<<<(BB * NN) / 256, 256, 0, stream>>>(semb, xx);
  dcp_sumsq<<<(BB * MM) / 256, 256, 0, stream>>>(temb, yyb);
  dcp_main<<<BB * (NN / 16), 128, 0, stream>>>(X16, Y16, xx, yyb, tgts, corr);
  dcp_finalize<<<1, 32, 0, stream>>>(srcs, corr, out);
}
